// SparseLayerAsEnsemble_39324720562890
// MI455X (gfx1250) — matrix operand order repaired and checked by reading, not compiled, not run on hardware
//
#include <hip/hip_runtime.h>

// ---------------------------------------------------------------------------
// Problem constants (from reference)
// ---------------------------------------------------------------------------
#define BATCH   16384
#define NIN     4096
#define NOUT    1024
#define NNZ     262144
#define BN_EPS  1e-3f

// GEMM tiling
#define BM 128          // rows per block
#define BNT 64          // cols per block
#define BK 32           // K per WMMA step (bf16 WMMA K)
#define LDA 40          // padded LDS pitch (bf16 elems) for A tile
#define LDB 40          // padded LDS pitch (bf16 elems) for B tile

typedef __attribute__((ext_vector_type(16))) __bf16 bf16x16;
typedef __attribute__((ext_vector_type(8)))  __bf16 bf16x8;
typedef __attribute__((ext_vector_type(8)))  float  f32x8;

// ---------------------------------------------------------------------------
// 1) zero the fp32 transposed-W scratch
// ---------------------------------------------------------------------------
__global__ void zero_f32_kernel(float* __restrict__ p, int n) {
    int i = blockIdx.x * blockDim.x + threadIdx.x;
    int stride = gridDim.x * blockDim.x;
    for (; i < n; i += stride) p[i] = 0.0f;
}

// ---------------------------------------------------------------------------
// 2) scatter sparse values into Wt[col][row] (fp32, duplicates accumulate)
// ---------------------------------------------------------------------------
__global__ void scatter_wt_kernel(const float* __restrict__ vals,
                                  const int* __restrict__ rows,
                                  const int* __restrict__ cols,
                                  float* __restrict__ wt) {
    int i = blockIdx.x * blockDim.x + threadIdx.x;
    if (i < NNZ) {
        atomicAdd(&wt[(size_t)cols[i] * NIN + rows[i]], vals[i]);
    }
}

// ---------------------------------------------------------------------------
// 3a) fold BN into per-k scale/bias:  h = x*s + b
// ---------------------------------------------------------------------------
__global__ void bn_params_kernel(const float* __restrict__ gamma,
                                 const float* __restrict__ beta,
                                 const float* __restrict__ mean,
                                 const float* __restrict__ var,
                                 float* __restrict__ s,
                                 float* __restrict__ b) {
    int i = blockIdx.x * blockDim.x + threadIdx.x;
    if (i < NIN) {
        float sc = gamma[i] * rsqrtf(var[i] + BN_EPS);
        s[i] = sc;
        b[i] = beta[i] - mean[i] * sc;
    }
}

// ---------------------------------------------------------------------------
// 3b) convert Wt fp32 -> bf16 once (read many times in GEMM)
// ---------------------------------------------------------------------------
__global__ void cvt_bf16_kernel(const float* __restrict__ src,
                                __bf16* __restrict__ dst, int n) {
    int i = blockIdx.x * blockDim.x + threadIdx.x;
    int stride = gridDim.x * blockDim.x;
    for (; i < n; i += stride) dst[i] = (__bf16)src[i];
}

// ---------------------------------------------------------------------------
// 4) Fused BN + bf16-WMMA GEMM:  out = (x*s+b) @ W,  W given as Wt[N][K] bf16
//    Block: 256 threads = 8 waves (wave32); block tile 128x64; wave tile 32x32
//    = 2x2 v_wmma_f32_16x16x32_bf16 accumulators.
//    Register-staged double buffering for x, Wt, scale and bias: all global
//    loads for tile k+1 are issued while tile k computes; the LDS staging
//    phase between barriers is pure VALU + ds_store (no global-load waits).
//    Grid: x = column blocks (16), y = row blocks (128) so the 16 blocks
//    sharing an x-stripe launch back-to-back and hit L2 for x.
// ---------------------------------------------------------------------------
__global__ __launch_bounds__(256)
void gemm_bn_wmma_kernel(const float* __restrict__ x,
                         const __bf16* __restrict__ wt,
                         const float* __restrict__ scale,
                         const float* __restrict__ bias,
                         float* __restrict__ out) {
    __shared__ __bf16 As[BM * LDA];   // [m][k] bf16, padded pitch
    __shared__ __bf16 Bs[BNT * LDB];  // [n][k] bf16 (W transposed), padded pitch

    const int tid   = threadIdx.x;
    const int lane  = tid & 31;
    const int wave  = tid >> 5;       // 0..7
    const int waveM = wave & 3;       // 0..3  -> 32-row slice
    const int waveN = wave >> 2;      // 0..1  -> 32-col slice
    const int cn    = blockIdx.x * BNT;   // column block (fast launch dim)
    const int rm    = blockIdx.y * BM;    // row block

    // A-tile loader mapping: 2 threads per row, 16 floats each
    const int ar = tid >> 1;          // 0..127
    const int ac = (tid & 1) * 16;    // 0 or 16
    // B-tile loader mapping: 4 threads per row, 8 bf16 each
    const int br = tid >> 2;          // 0..63
    const int bc = (tid & 3) * 8;     // 0,8,16,24

    // WMMA fragment lane decomposition (per ISA 16-bit layouts)
    const int frow = lane & 15;       // M (for A) / N (for B)
    const int ksel = lane >> 4;       // selects K half-chunks

    f32x8 acc[2][2] = {};

    const float*  xrow = x + (size_t)(rm + ar) * NIN;
    const __bf16* wrow = wt + (size_t)(cn + br) * NIN;

    // register staging for double buffering (x, scale, bias, Wt)
    float4 xr[4], sr[4], brg[4];
    bf16x8 wr;

    // prologue: fetch tile 0
    {
        const float4* xp = (const float4*)(xrow + ac);
        const float4* sp = (const float4*)(scale + ac);
        const float4* bp = (const float4*)(bias + ac);
#pragma unroll
        for (int q = 0; q < 4; ++q) { xr[q] = xp[q]; sr[q] = sp[q]; brg[q] = bp[q]; }
        wr = *(const bf16x8*)(wrow + bc);
    }

    for (int k0 = 0; k0 < NIN; k0 += BK) {
        // -- stage current tile from registers into LDS (BN fused, fp32->bf16)
        //    pure VALU + ds_store: no global-load dependency here
        {
#pragma unroll
            for (int h = 0; h < 2; ++h) {
                bf16x8 ch;
#pragma unroll
                for (int q = 0; q < 2; ++q) {
                    float4 v  = xr[2 * h + q];
                    float4 sv = sr[2 * h + q];
                    float4 bv = brg[2 * h + q];
                    ch[4 * q + 0] = (__bf16)fmaf(v.x, sv.x, bv.x);
                    ch[4 * q + 1] = (__bf16)fmaf(v.y, sv.y, bv.y);
                    ch[4 * q + 2] = (__bf16)fmaf(v.z, sv.z, bv.z);
                    ch[4 * q + 3] = (__bf16)fmaf(v.w, sv.w, bv.w);
                }
                *(bf16x8*)&As[ar * LDA + ac + 8 * h] = ch;   // ds_store_b128
            }
            *(bf16x8*)&Bs[br * LDB + bc] = wr;               // ds_store_b128
        }
        __syncthreads();

        // -- issue next tile's global loads (overlap with ds_load + WMMA) --
        const int kn = k0 + BK;
        if (kn < NIN) {
            const float4* xp = (const float4*)(xrow + kn + ac);
            const float4* sp = (const float4*)(scale + kn + ac);
            const float4* bp = (const float4*)(bias + kn + ac);
#pragma unroll
            for (int q = 0; q < 4; ++q) { xr[q] = xp[q]; sr[q] = sp[q]; brg[q] = bp[q]; }
            wr = *(const bf16x8*)(wrow + kn + bc);
            if (kn + BK < NIN) {
                __builtin_prefetch(xrow + kn + BK + ac, 0, 3);  // near-cache prefetch
            }
        }

        // -- build fragments from LDS --
        // A 16x32 bf16: lane<16 -> K 0..7 & 16..23 ; lane>=16 -> K 8..15 & 24..31
        bf16x16 aF[2];
#pragma unroll
        for (int i = 0; i < 2; ++i) {
            int row = waveM * 32 + i * 16 + frow;
            bf16x8 c0 = *(const bf16x8*)&As[row * LDA + ksel * 8];
            bf16x8 c1 = *(const bf16x8*)&As[row * LDA + 16 + ksel * 8];
            aF[i] = __builtin_shufflevector(c0, c1, 0, 1, 2, 3, 4, 5, 6, 7,
                                            8, 9, 10, 11, 12, 13, 14, 15);
        }
        // B 32x16 bf16 from Bs[n][k]: lane<16 -> K 0..7 & 16..23 ; lane>=16 -> K 8..15 & 24..31
        bf16x16 bF[2];
#pragma unroll
        for (int j = 0; j < 2; ++j) {
            int row = waveN * 32 + j * 16 + frow;
            bf16x8 c0 = *(const bf16x8*)&Bs[row * LDB + ksel * 8];
            bf16x8 c1 = *(const bf16x8*)&Bs[row * LDB + 16 + ksel * 8];
            bF[j] = __builtin_shufflevector(c0, c1, 0, 1, 2, 3, 4, 5, 6, 7,
                                            8, 9, 10, 11, 12, 13, 14, 15);
        }

        // -- 4x v_wmma_f32_16x16x32_bf16 --
#pragma unroll
        for (int i = 0; i < 2; ++i)
#pragma unroll
            for (int j = 0; j < 2; ++j)
                acc[i][j] = __builtin_amdgcn_wmma_f32_16x16x32_bf16(
                    false, aF[i], false, bF[j], (short)0, acc[i][j], false, false);

        __syncthreads();
    }

    // -- write out: C/D layout: lane<16 -> M 0..7, lane>=16 -> M 8..15; N=lane&15
    // out is streaming (never re-read): store non-temporally to spare L2 for x.
    const int ncol  = lane & 15;
    const int mbase = (lane >> 4) * 8;
#pragma unroll
    for (int i = 0; i < 2; ++i)
#pragma unroll
        for (int j = 0; j < 2; ++j) {
            int col = cn + waveN * 32 + j * 16 + ncol;
#pragma unroll
            for (int r = 0; r < 8; ++r) {
                int row = rm + waveM * 32 + i * 16 + mbase + r;
                __builtin_nontemporal_store(acc[i][j][r], &out[(size_t)row * NOUT + col]);
            }
        }
}

// ---------------------------------------------------------------------------
// Launch: inputs in setup_inputs() order:
// 0:x 1:sp_values 2:gamma 3:beta 4:moving_mean 5:moving_var 6:sp_rows 7:sp_cols
// ---------------------------------------------------------------------------
extern "C" void kernel_launch(void* const* d_in, const int* in_sizes, int n_in,
                              void* d_out, int out_size, void* d_ws, size_t ws_size,
                              hipStream_t stream) {
    (void)in_sizes; (void)n_in; (void)out_size; (void)ws_size;

    const float* x      = (const float*)d_in[0];
    const float* vals   = (const float*)d_in[1];
    const float* gamma  = (const float*)d_in[2];
    const float* beta   = (const float*)d_in[3];
    const float* mean   = (const float*)d_in[4];
    const float* var    = (const float*)d_in[5];
    const int*   rows   = (const int*)d_in[6];
    const int*   cols   = (const int*)d_in[7];
    float*       out    = (float*)d_out;

    // workspace layout
    char* ws = (char*)d_ws;
    float*  wtf   = (float*)ws;                                          // 16 MB  Wt fp32 [NOUT][NIN]
    __bf16* wtb   = (__bf16*)(ws + (size_t)NOUT * NIN * sizeof(float));  // 8 MB   Wt bf16
    float*  scale = (float*)(ws + (size_t)NOUT * NIN * 6);               // 16 KB
    float*  bias  = scale + NIN;                                         // 16 KB

    const int nW = NOUT * NIN;

    zero_f32_kernel<<<4096, 256, 0, stream>>>(wtf, nW);
    scatter_wt_kernel<<<NNZ / 256, 256, 0, stream>>>(vals, rows, cols, wtf);
    bn_params_kernel<<<NIN / 256, 256, 0, stream>>>(gamma, beta, mean, var, scale, bias);
    cvt_bf16_kernel<<<4096, 256, 0, stream>>>(wtf, wtb, nW);

    // x = column blocks (16, fast dim -> L2 reuse of x stripes), y = row blocks
    dim3 grid(NOUT / BNT, BATCH / BM);   // 16 x 128
    gemm_bn_wmma_kernel<<<grid, 256, 0, stream>>>(x, wtb, scale, bias, out);
}
